// pose_prediction_89584427860623
// MI455X (gfx1250) — compile-verified, matrix operand order checked by reading, not run on hardware
//
#include <hip/hip_runtime.h>
#include <hip/hip_bf16.h>

typedef __attribute__((ext_vector_type(16))) __bf16 bf16x16;
typedef __attribute__((ext_vector_type(8)))  float  v8f;

#define B_ROWS 16384
#define DIN    72
#define KPAD1  96
#define HID    256
#define PD     144
#define PDPAD  160
#define CD     6890
#define CDPAD  6912

__device__ __forceinline__ unsigned short f2bf(float f) {
  unsigned int u = __float_as_uint(f);
  unsigned int r = u + 0x7FFFu + ((u >> 16) & 1u);   // round-to-nearest-even
  return (unsigned short)(r >> 16);
}

// fp32 -> bf16 with zero padding of rows and cols (K padded to mult of 32,
// weight rows padded to mult of 32 so every WMMA tile reads valid zeros).
__global__ void cvt_bf16_pad(const float* __restrict__ src, unsigned short* __restrict__ dst,
                             int srcRows, int srcCols, int dstRows, int dstCols) {
  long long total = (long long)dstRows * dstCols;
  for (long long i = (long long)blockIdx.x * blockDim.x + threadIdx.x; i < total;
       i += (long long)gridDim.x * blockDim.x) {
    int r = (int)(i / dstCols);
    int c = (int)(i - (long long)r * dstCols);
    float v = (r < srcRows && c < srcCols) ? src[(long long)r * srcCols + c] : 0.0f;
    dst[i] = f2bf(v);
  }
}

union FragU {
  bf16x16 v;
  uint4   q[2];
};

__device__ __forceinline__ v8f wmma_bf16(const FragU& a, const FragU& b, v8f c) {
  return __builtin_amdgcn_wmma_f32_16x16x32_bf16(false, a.v, false, b.v, (short)0, c,
                                                 false, false);
}

template <int OUT_BF16, int ACT>
__device__ __forceinline__ void store_tile(const v8f& acc, void* __restrict__ outv,
                                           int row0, int n, int Nvalid, int hi,
                                           const float* __restrict__ bias) {
  if (n >= Nvalid) return;               // drop padding columns
  const float bv = bias[n];
#pragma unroll
  for (int r = 0; r < 8; ++r) {
    const int row = row0 + r + hi * 8;   // D layout: VGPR r -> row r (+8 for hi half)
    float v = acc[r] + bv;
    if (ACT == 1) v = fmaxf(v, 0.0f);
    if (ACT == 2) v = 1.0f / (1.0f + expf(-v));
    if (OUT_BF16) ((unsigned short*)outv)[(size_t)row * Nvalid + n] = f2bf(v);
    else          ((float*)outv)[(size_t)row * Nvalid + n] = v;
  }
}

// Register-blocked WMMA GEMM: each wave computes a 32(M)x32(N) macro-tile as
// 2x2 WMMA tiles, reusing each A/B fragment twice -> 2 b128 loads per WMMA.
// Block = 4 waves covering 32(M) x 128(N). A, W are bf16 row-major, K%32==0,
// Npad%32==0. D = act(A[M,K] * W[N,K]^T + bias).
// ISA 16-bit fragment layout: lanes 0-15 -> row=lane, K=0..7 & 16..23;
// lanes 16-31 -> row=lane-16, K=8..15 & 24..31 (B^T == W loads like A).
template <int OUT_BF16, int ACT>
__global__ void gemm_wmma_bf16(const unsigned short* __restrict__ A,
                               const unsigned short* __restrict__ W,
                               const float* __restrict__ bias,
                               void* __restrict__ outv,
                               int Npad, int K, int Nvalid) {
  const int lane = threadIdx.x & 31;
  const int wave = threadIdx.x >> 5;
  const int n0 = blockIdx.x * 128 + wave * 32;
  if (n0 >= Npad) return;
  const int m0 = blockIdx.y * 32;
  const int idx = lane & 15;
  const int hi  = lane >> 4;

  const unsigned short* aR0 = A + (size_t)(m0 + idx) * K + hi * 8;
  const unsigned short* aR1 = aR0 + (size_t)16 * K;
  const unsigned short* wR0 = W + (size_t)(n0 + idx) * K + hi * 8;
  const unsigned short* wR1 = wR0 + (size_t)16 * K;

  v8f acc00 = {}, acc01 = {}, acc10 = {}, acc11 = {};
  for (int k = 0; k < K; k += 32) {
    FragU fa0, fa1, fb0, fb1;
    fa0.q[0] = *(const uint4*)(aR0 + k);
    fa0.q[1] = *(const uint4*)(aR0 + k + 16);
    fa1.q[0] = *(const uint4*)(aR1 + k);
    fa1.q[1] = *(const uint4*)(aR1 + k + 16);
    fb0.q[0] = *(const uint4*)(wR0 + k);
    fb0.q[1] = *(const uint4*)(wR0 + k + 16);
    fb1.q[0] = *(const uint4*)(wR1 + k);
    fb1.q[1] = *(const uint4*)(wR1 + k + 16);
    acc00 = wmma_bf16(fa0, fb0, acc00);
    acc01 = wmma_bf16(fa0, fb1, acc01);
    acc10 = wmma_bf16(fa1, fb0, acc10);
    acc11 = wmma_bf16(fa1, fb1, acc11);
  }

  const int nA = n0 + idx;
  const int nB = n0 + 16 + idx;
  store_tile<OUT_BF16, ACT>(acc00, outv, m0,      nA, Nvalid, hi, bias);
  store_tile<OUT_BF16, ACT>(acc01, outv, m0,      nB, Nvalid, hi, bias);
  store_tile<OUT_BF16, ACT>(acc10, outv, m0 + 16, nA, Nvalid, hi, bias);
  store_tile<OUT_BF16, ACT>(acc11, outv, m0 + 16, nB, Nvalid, hi, bias);
}

// rot6d -> rotmat -> axis-angle, one thread per joint (16384*24 joints).
__global__ void rot6d_to_aa(const float* __restrict__ pose6, float* __restrict__ out) {
  const int total = B_ROWS * 24;
  int i = blockIdx.x * blockDim.x + threadIdx.x;
  if (i >= total) return;
  const int s = i / 24, j = i - s * 24;
  const float* p = pose6 + (size_t)s * PD + j * 6;
  float a1x = p[0], a1y = p[1], a1z = p[2];
  float a2x = p[3], a2y = p[4], a2z = p[5];

  float n1 = fmaxf(sqrtf(a1x*a1x + a1y*a1y + a1z*a1z), 1e-12f);
  float b1x = a1x/n1, b1y = a1y/n1, b1z = a1z/n1;
  float d = b1x*a2x + b1y*a2y + b1z*a2z;
  float px = a2x - d*b1x, py = a2y - d*b1y, pz = a2z - d*b1z;
  float n2 = fmaxf(sqrtf(px*px + py*py + pz*pz), 1e-12f);
  float b2x = px/n2, b2y = py/n2, b2z = pz/n2;
  float b3x = b1y*b2z - b1z*b2y;
  float b3y = b1z*b2x - b1x*b2z;
  float b3z = b1x*b2y - b1y*b2x;
  // R = [b1 | b2 | b3] (columns)
  const float R00=b1x, R01=b2x, R02=b3x;
  const float R10=b1y, R11=b2y, R12=b3y;
  const float R20=b1z, R21=b2z, R22=b3z;

  float tr = R00 + R11 + R22;
  float c = fminf(fmaxf((tr - 1.0f) * 0.5f, -1.0f), 1.0f);
  float theta = acosf(c);
  float vx = R21 - R12, vy = R02 - R20, vz = R10 - R01;
  float sn = sinf(theta);
  float factor = (sn > 1e-6f) ? theta / (2.0f * fmaxf(sn, 1e-6f)) : 0.5f;
  float gx = vx*factor, gy = vy*factor, gz = vz*factor;

  float a0 = sqrtf(fmaxf((R00 + 1.0f) * 0.5f, 0.0f));
  float a1 = sqrtf(fmaxf((R11 + 1.0f) * 0.5f, 0.0f));
  float a2 = sqrtf(fmaxf((R22 + 1.0f) * 0.5f, 0.0f));
  int k = 0; float dm = R00;
  if (R11 > dm) { dm = R11; k = 1; }
  if (R22 > dm) { dm = R22; k = 2; }
  float r0, r1, r2;  // row k of Msym = (R+R^T)*0.25 + 0.5I
  if (k == 0)      { r0 = R00*0.5f + 0.5f;      r1 = (R01+R10)*0.25f;     r2 = (R02+R20)*0.25f; }
  else if (k == 1) { r0 = (R10+R01)*0.25f;      r1 = R11*0.5f + 0.5f;     r2 = (R12+R21)*0.25f; }
  else             { r0 = (R20+R02)*0.25f;      r1 = (R21+R12)*0.25f;     r2 = R22*0.5f + 0.5f; }
  float p0 = a0 * ((r0 >= 0.0f) ? 1.0f : -1.0f) * theta;
  float p1 = a1 * ((r1 >= 0.0f) ? 1.0f : -1.0f) * theta;
  float p2 = a2 * ((r2 >= 0.0f) ? 1.0f : -1.0f) * theta;
  bool near_pi = c < (-1.0f + 1e-4f);

  float* o = out + (size_t)s * 72 + j * 3;
  o[0] = near_pi ? p0 : gx;
  o[1] = near_pi ? p1 : gy;
  o[2] = near_pi ? p2 : gz;
}

extern "C" void kernel_launch(void* const* d_in, const int* in_sizes, int n_in,
                              void* d_out, int out_size, void* d_ws, size_t ws_size,
                              hipStream_t stream) {
  const float* x  = (const float*)d_in[0];
  const float* W1 = (const float*)d_in[1];
  const float* b1 = (const float*)d_in[2];
  const float* W2 = (const float*)d_in[3];
  const float* b2 = (const float*)d_in[4];
  const float* W3 = (const float*)d_in[5];
  const float* b3 = (const float*)d_in[6];
  const float* Wp = (const float*)d_in[7];
  const float* bp = (const float*)d_in[8];
  const float* Wc = (const float*)d_in[9];
  const float* bc = (const float*)d_in[10];

  char* ws = (char*)d_ws;
  size_t off = 0;
  auto alloc = [&](size_t bytes) -> void* {
    void* p = ws + off;
    off = (off + bytes + 255) & ~(size_t)255;
    return p;
  };
  unsigned short* xb  = (unsigned short*)alloc((size_t)B_ROWS * KPAD1 * 2);
  unsigned short* W1b = (unsigned short*)alloc((size_t)HID * KPAD1 * 2);
  unsigned short* W2b = (unsigned short*)alloc((size_t)HID * HID * 2);
  unsigned short* W3b = (unsigned short*)alloc((size_t)HID * HID * 2);
  unsigned short* Wpb = (unsigned short*)alloc((size_t)PDPAD * HID * 2);
  unsigned short* Wcb = (unsigned short*)alloc((size_t)CDPAD * HID * 2);
  unsigned short* h1b = (unsigned short*)alloc((size_t)B_ROWS * HID * 2);
  unsigned short* h2b = (unsigned short*)alloc((size_t)B_ROWS * HID * 2);
  unsigned short* h3b = (unsigned short*)alloc((size_t)B_ROWS * HID * 2);
  float*          p6  = (float*)alloc((size_t)B_ROWS * PD * 4);

  auto cvt = [&](const float* src, unsigned short* dst, int sr, int sc, int dr, int dc) {
    long long tot = (long long)dr * dc;
    int blocks = (int)((tot + 255) / 256);
    if (blocks > 4096) blocks = 4096;
    cvt_bf16_pad<<<blocks, 256, 0, stream>>>(src, dst, sr, sc, dr, dc);
  };
  cvt(x,  xb,  B_ROWS, DIN, B_ROWS, KPAD1);
  cvt(W1, W1b, HID,    DIN, HID,    KPAD1);
  cvt(W2, W2b, HID,    HID, HID,    HID);
  cvt(W3, W3b, HID,    HID, HID,    HID);
  cvt(Wp, Wpb, PD,     HID, PDPAD,  HID);
  cvt(Wc, Wcb, CD,     HID, CDPAD,  HID);

  const dim3 blk(128);                 // 4 waves; each wave: 32x32 macro-tile
  const int mBlocks = B_ROWS / 32;     // 512

  // h1 = relu(x @ W1^T + b1)          N=256 -> grid.x = 256/128 = 2
  gemm_wmma_bf16<1, 1><<<dim3(2, mBlocks), blk, 0, stream>>>(xb,  W1b, b1, h1b, HID, KPAD1, HID);
  // h2 = relu(h1 @ W2^T + b2)
  gemm_wmma_bf16<1, 1><<<dim3(2, mBlocks), blk, 0, stream>>>(h1b, W2b, b2, h2b, HID, HID, HID);
  // h3 = relu(h2 @ W3^T + b3)
  gemm_wmma_bf16<1, 1><<<dim3(2, mBlocks), blk, 0, stream>>>(h2b, W3b, b3, h3b, HID, HID, HID);
  // pose6 = h3 @ Wp^T + bp            Npad=160 -> grid.x = ceil(160/128) = 2
  gemm_wmma_bf16<0, 0><<<dim3(2, mBlocks), blk, 0, stream>>>(h3b, Wpb, bp, p6, PDPAD, HID, PD);
  // pred_contact = sigmoid(h3 @ Wc^T + bc)  Npad=6912 -> grid.x = 6912/128 = 54
  float* contact = (float*)d_out + (size_t)B_ROWS * 72;
  gemm_wmma_bf16<0, 2><<<dim3(54, mBlocks), blk, 0, stream>>>(h3b, Wcb, bc, contact, CDPAD, HID, CD);

  // pred_pose: SO(3) log map over 16384*24 joints
  rot6d_to_aa<<<(B_ROWS * 24 + 255) / 256, 256, 0, stream>>>(p6, (float*)d_out);
}
